// BasicBlock_85761906966887
// MI455X (gfx1250) — matrix operand order repaired and checked by reading, not compiled, not run on hardware
//
#include <hip/hip_runtime.h>
#include <hip/hip_bf16.h>

#define N_PTS 131072
#define C_CH  96
#define KOFF  27
#define EPSV  1e-5f
#define SLOPE 0.01f

typedef __attribute__((ext_vector_type(16))) _Float16 v16h;
typedef __attribute__((ext_vector_type(8)))  float    v8f;
typedef int v4i __attribute__((vector_size(16)));

#if defined(__gfx1250__) && __has_builtin(__builtin_amdgcn_global_load_async_to_lds_b128)
#define HAVE_ASYNC_LDS 1
#else
#define HAVE_ASYNC_LDS 0
#endif

// 16-byte global -> LDS mover (async on gfx1250 when available)
__device__ __forceinline__ void ldsmove_b128(const void* g, void* l) {
#if HAVE_ASYNC_LDS
    __builtin_amdgcn_global_load_async_to_lds_b128((v4i*)g, (v4i*)l, 0, 0);
#else
    *(uint4*)l = *(const uint4*)g;
#endif
}

__device__ __forceinline__ void ldsmove_wait() {
#if HAVE_ASYNC_LDS
#if __has_builtin(__builtin_amdgcn_s_wait_asynccnt)
    __builtin_amdgcn_s_wait_asynccnt(0);
#else
    asm volatile("s_wait_asynccnt 0x0" ::: "memory");
#endif
#endif
}

// ---------------- prep: x fp32 -> f16 (padded zero row) ----------------
__global__ __launch_bounds__(256) void prep_x_kernel(const float* __restrict__ x,
                                                     _Float16* __restrict__ xb) {
    size_t idx = (size_t)blockIdx.x * 256 + threadIdx.x;
    size_t total = (size_t)(N_PTS + 1) * C_CH;
    if (idx >= total) return;
    xb[idx] = (idx < (size_t)N_PTS * C_CH) ? (_Float16)x[idx] : (_Float16)0.0f;
}

// ---------------- prep: W fp32 [K][Cin][Cout] -> f16 transposed [K][Cout][Cin] ----------------
__global__ __launch_bounds__(256) void prep_w_kernel(const float* __restrict__ W1,
                                                     const float* __restrict__ W2,
                                                     _Float16* __restrict__ w1t,
                                                     _Float16* __restrict__ w2t) {
    int idx = blockIdx.x * 256 + threadIdx.x;
    if (idx >= KOFF * C_CH * C_CH) return;
    int k  = idx / (C_CH * C_CH);
    int r  = idx % (C_CH * C_CH);
    int ci = r / C_CH;
    int co = r % C_CH;
    int dst = k * C_CH * C_CH + co * C_CH + ci;
    w1t[dst] = (_Float16)W1[idx];
    w2t[dst] = (_Float16)W2[idx];
}

// ---------------- sparse conv: async gather -> LDS (double buffered) -> WMMA f16 ----------------
// Block: 256 threads (8 waves), 128 rows per block. Per wave: 16 rows x 96 cols,
// acc = 6 tiles of v8f held across the K=27 offset loop.
__global__ __launch_bounds__(256) void sparse_conv_wmma_kernel(
    const unsigned int* __restrict__ featu,   // (N+1) rows x 48 dwords of f16
    const unsigned int* __restrict__ wtu,     // [K][96][96] f16 as dwords (4608/k)
    const long long*    __restrict__ nbr,     // [K][N]
    float*              __restrict__ out)     // [N][96] fp32
{
    __shared__ __align__(16) unsigned int Atile[2][128 * 48];  // 128 rows x 96 f16
    __shared__ __align__(16) unsigned int Wtile[2][96 * 48];   // 96 cout x 96 cin f16

    const int t      = threadIdx.x;
    const int lane   = t & 31;
    const int wv     = t >> 5;          // wave 0..7
    const int laneM  = lane & 15;
    const int laneHi = lane >> 4;       // 0 or 1
    const int rowbase = blockIdx.x * 128;

    v8f acc[6] = {};

    const int gr   = t & 127;           // gather row within tile
    const int ghlf = t >> 7;            // which half of the row this thread copies

    // stage offsets k into LDS buffer `buf` (16B chunks, async when available)
    auto stage = [&](int k, int buf) {
        // W[k]^T tile: 18432 B = 1152 x b128
        const char* wg = (const char*)(wtu + k * 4608);
        char*       wl = (char*)&Wtile[buf][0];
        for (int i = t; i < 1152; i += 256)
            ldsmove_b128(wg + i * 16, wl + i * 16);
        // gather 128 rows: 192 B each, 2 threads/row x 6 b128
        long long idxq = nbr[(long long)k * N_PTS + (rowbase + gr)];
        int src = (idxq >= 0 && idxq < N_PTS) ? (int)idxq : N_PTS;  // zero row
        const char* srow = (const char*)(featu + (size_t)src * 48) + ghlf * 96;
        char*       drow = (char*)&Atile[buf][0] + gr * 192 + ghlf * 96;
        #pragma unroll
        for (int j = 0; j < 6; ++j)
            ldsmove_b128(srow + j * 16, drow + j * 16);
    };

    stage(0, 0);

    #pragma unroll 1
    for (int k = 0; k < KOFF; ++k) {
        const int cur = k & 1;
        ldsmove_wait();       // my buffer-k chunks have landed in LDS
        __syncthreads();      // everyone's buffer-k chunks have landed
        if (k + 1 < KOFF)
            stage(k + 1, cur ^ 1);   // overlaps with the WMMA work below

        const unsigned int* arow = &Atile[cur][(wv * 16 + laneM) * 48];
        #pragma unroll
        for (int kc = 0; kc < 3; ++kc) {
            const int kb = kc * 16;     // dword base of this 32-wide K chunk
            union { v16h v; unsigned int u[8]; } A;
            // A f16 16x32, contiguous k order used here: pair j -> K = 16*laneHi + 2j, 2j+1 (matches B)
            #pragma unroll
            for (int j = 0; j < 8; ++j) {
                A.u[j] = arow[kb + laneHi * 8 + j];
            }
            #pragma unroll
            for (int ct = 0; ct < 6; ++ct) {
                const unsigned int* brow = &Wtile[cur][(ct * 16 + laneM) * 48];
                union { v16h v; unsigned int u[8]; } B;
                // B 32x16, contiguous k order used here: lane -> column, halves K = 16*laneHi + 2j,2j+1
                #pragma unroll
                for (int j = 0; j < 8; ++j)
                    B.u[j] = brow[kb + laneHi * 8 + j];
                acc[ct] = __builtin_amdgcn_wmma_f32_16x16x32_f16(
                    false, A.v, false, B.v, (short)0, acc[ct], false, false);
            }
        }
        __syncthreads();      // all waves done reading buffer k before it is restaged
    }

    // C/D layout: VGPR v -> M = v + 8*laneHi, N = lane&15
    #pragma unroll
    for (int ct = 0; ct < 6; ++ct) {
        int col = ct * 16 + laneM;
        #pragma unroll
        for (int v = 0; v < 8; ++v) {
            int m = v + laneHi * 8;
            out[(size_t)(rowbase + wv * 16 + m) * C_CH + col] = acc[ct][v];
        }
    }
}

// ---------------- deterministic per-channel sum / sumsq ----------------
__global__ __launch_bounds__(256) void reduce_stats_kernel(const float* __restrict__ h,
                                                           float* __restrict__ sum,
                                                           float* __restrict__ sq) {
    __shared__ float s1[256];
    __shared__ float s2[256];
    const int c = blockIdx.x;
    float a = 0.f, b = 0.f;
    for (int n = threadIdx.x; n < N_PTS; n += 256) {
        float v = h[(size_t)n * C_CH + c];
        a += v; b += v * v;
    }
    s1[threadIdx.x] = a; s2[threadIdx.x] = b;
    __syncthreads();
    for (int sft = 128; sft > 0; sft >>= 1) {
        if (threadIdx.x < sft) {
            s1[threadIdx.x] += s1[threadIdx.x + sft];
            s2[threadIdx.x] += s2[threadIdx.x + sft];
        }
        __syncthreads();
    }
    if (threadIdx.x == 0) { sum[c] = s1[0]; sq[c] = s2[0]; }
}

// ---------------- instance-norm + LeakyReLU -> f16 (padded) ----------------
__global__ __launch_bounds__(256) void norm1_kernel(const float* __restrict__ h,
                                                    const float* __restrict__ sum,
                                                    const float* __restrict__ sq,
                                                    const float* __restrict__ gamma,
                                                    const float* __restrict__ beta,
                                                    _Float16* __restrict__ outh) {
    size_t idx = (size_t)blockIdx.x * 256 + threadIdx.x;
    size_t total = (size_t)(N_PTS + 1) * C_CH;
    if (idx >= total) return;
    if (idx >= (size_t)N_PTS * C_CH) { outh[idx] = (_Float16)0.0f; return; }
    int c = (int)(idx % C_CH);
    float mean = sum[c] * (1.0f / N_PTS);
    float var  = sq[c] * (1.0f / N_PTS) - mean * mean;
    float rstd = rsqrtf(var + EPSV);
    float v = (h[idx] - mean) * rstd * gamma[c] + beta[c];
    v = (v >= 0.f) ? v : SLOPE * v;
    outh[idx] = (_Float16)v;
}

// ---------------- norm2 + residual + LeakyReLU (in place on d_out) ----------------
__global__ __launch_bounds__(256) void finalize_kernel(float* __restrict__ o2,
                                                       const float* __restrict__ x,
                                                       const float* __restrict__ sum,
                                                       const float* __restrict__ sq,
                                                       const float* __restrict__ gamma,
                                                       const float* __restrict__ beta) {
    size_t idx = (size_t)blockIdx.x * 256 + threadIdx.x;
    if (idx >= (size_t)N_PTS * C_CH) return;
    int c = (int)(idx % C_CH);
    float mean = sum[c] * (1.0f / N_PTS);
    float var  = sq[c] * (1.0f / N_PTS) - mean * mean;
    float rstd = rsqrtf(var + EPSV);
    float v = (o2[idx] - mean) * rstd * gamma[c] + beta[c] + x[idx];
    o2[idx] = (v >= 0.f) ? v : SLOPE * v;
}

extern "C" void kernel_launch(void* const* d_in, const int* in_sizes, int n_in,
                              void* d_out, int out_size, void* d_ws, size_t ws_size,
                              hipStream_t stream) {
    const float*     x      = (const float*)d_in[0];
    const float*     W1     = (const float*)d_in[1];
    const float*     W2     = (const float*)d_in[2];
    const float*     gamma1 = (const float*)d_in[3];
    const float*     beta1  = (const float*)d_in[4];
    const float*     gamma2 = (const float*)d_in[5];
    const float*     beta2  = (const float*)d_in[6];
    const long long* nbr1   = (const long long*)d_in[7];
    const long long* nbr2   = (const long long*)d_in[8];
    float* out = (float*)d_out;

    // workspace layout (256B aligned)
    char* ws = (char*)d_ws;
    const size_t XB_BYTES = ((size_t)(N_PTS + 1) * C_CH * 2 + 255) & ~(size_t)255; // f16 feat (padded)
    const size_t WT_BYTES = ((size_t)KOFF * C_CH * C_CH * 2 + 255) & ~(size_t)255; // f16 weights^T
    const size_t O1_BYTES = ((size_t)N_PTS * C_CH * 4 + 255) & ~(size_t)255;       // fp32 conv1 out
    _Float16* xb    = (_Float16*)(ws);
    _Float16* h1b   = (_Float16*)(ws + XB_BYTES);
    _Float16* w1t   = (_Float16*)(ws + 2 * XB_BYTES);
    _Float16* w2t   = (_Float16*)(ws + 2 * XB_BYTES + WT_BYTES);
    float*    out1  = (float*)   (ws + 2 * XB_BYTES + 2 * WT_BYTES);
    float*    stats = (float*)   (ws + 2 * XB_BYTES + 2 * WT_BYTES + O1_BYTES);
    float* sum1 = stats, *sq1 = stats + 96, *sum2 = stats + 192, *sq2 = stats + 288;

    const size_t padElems = (size_t)(N_PTS + 1) * C_CH;
    const int padBlocks  = (int)((padElems + 255) / 256);
    const int wBlocks    = (KOFF * C_CH * C_CH + 255) / 256;
    const int convBlocks = N_PTS / 128;
    const int elemBlocks = (int)(((size_t)N_PTS * C_CH + 255) / 256);

    prep_x_kernel<<<padBlocks, 256, 0, stream>>>(x, xb);
    prep_w_kernel<<<wBlocks, 256, 0, stream>>>(W1, W2, w1t, w2t);

    sparse_conv_wmma_kernel<<<convBlocks, 256, 0, stream>>>(
        (const unsigned int*)xb, (const unsigned int*)w1t, nbr1, out1);
    reduce_stats_kernel<<<C_CH, 256, 0, stream>>>(out1, sum1, sq1);
    norm1_kernel<<<padBlocks, 256, 0, stream>>>(out1, sum1, sq1, gamma1, beta1, h1b);

    sparse_conv_wmma_kernel<<<convBlocks, 256, 0, stream>>>(
        (const unsigned int*)h1b, (const unsigned int*)w2t, nbr2, out);
    reduce_stats_kernel<<<C_CH, 256, 0, stream>>>(out, sum2, sq2);
    finalize_kernel<<<elemBlocks, 256, 0, stream>>>(out, x, sum2, sq2, gamma2, beta2);
}
